// GCNRegressor_18107582119953
// MI455X (gfx1250) — compile-verified, matrix operand order checked by reading, not run on hardware
//
#include <hip/hip_runtime.h>
#include <hip/hip_bf16.h>

#define N_NODES  50000
#define N_EDGES  800000
#define N_GRAPHS 64

typedef __attribute__((ext_vector_type(16))) __bf16         v16bf;
typedef __attribute__((ext_vector_type(8)))  float          v8f;
typedef __attribute__((ext_vector_type(16))) unsigned short v16us;
typedef __attribute__((ext_vector_type(8)))  unsigned short v8us;

typedef __attribute__((address_space(3))) unsigned short lds_us;

__device__ __forceinline__ unsigned short f2bf(float f) {
  unsigned u = __float_as_uint(f);
  u += 0x7FFFu + ((u >> 16) & 1u);   // round-to-nearest-even
  return (unsigned short)(u >> 16);
}

// ---------------- utility kernels ----------------
__global__ void fill_f32(float* __restrict__ p, float v, int n) {
  int i = blockIdx.x * blockDim.x + threadIdx.x;
  if (i < n) p[i] = v;
}

__global__ void deg_acc(const int* __restrict__ dst, const float* __restrict__ ew,
                        float* __restrict__ deg, int e) {
  int i = blockIdx.x * blockDim.x + threadIdx.x;
  if (i < e) atomicAdd(&deg[dst[i]], ew[i]);
}

__global__ void rsqrt_ip(float* __restrict__ p, int n) {
  int i = blockIdx.x * blockDim.x + threadIdx.x;
  if (i < n) p[i] = rsqrtf(p[i]);   // deg >= 1 always (self loop)
}

__global__ void norm_k(const int* __restrict__ src, const int* __restrict__ dst,
                       const float* __restrict__ ew, const float* __restrict__ dinv,
                       float* __restrict__ nrm, int e) {
  int i = blockIdx.x * blockDim.x + threadIdx.x;
  if (i < e) nrm[i] = dinv[src[i]] * ew[i] * dinv[dst[i]];
}

// W [K x N] fp32 -> Wt [N x K] bf16 (transposed so K is contiguous per output column)
__global__ void conv_wt(const float* __restrict__ W, unsigned short* __restrict__ Wt,
                        int K, int N) {
  int i = blockIdx.x * blockDim.x + threadIdx.x;
  if (i >= K * N) return;
  int k = i / N, n = i - k * N;
  Wt[(size_t)n * K + k] = f2bf(W[i]);
}

// ---------------- strip-mined WMMA GEMM, async-LDS-staged B, fused epilogue ----------------
// One launch covers 128 output columns (col0 .. col0+127). Each wave computes a 16-row strip
// across those 128 columns (8 accumulator tiles = 64 VGPRs).
//   XW[m][col]  = relu?(A[m][:]) @ W[:, col]
//   OUT[m][col] = dinv[m]^2 * XW[m][col] + bias[col]     (self-loop + bias)
// Bts points at rows col0..col0+127 of the bf16-transposed weight (each row = K halfwords).
template <int K, int NTOT>
__global__ __launch_bounds__(128, 1)
void gemm_wmma_strip(const float* __restrict__ A, const unsigned short* __restrict__ Bts,
                     float* __restrict__ XW, float* __restrict__ OUT,
                     const float* __restrict__ dinv, const float* __restrict__ bias,
                     int col0, int relu_in) {
  constexpr int KC = 64;                  // k-chunk staged in LDS
  constexpr int KP = KC + 8;              // padded row stride (halfwords); 144B = 9*16B: aligned
  __shared__ unsigned short sB[128 * KP]; // 18 KB

  const int lane = threadIdx.x & 31;
  int tr = blockIdx.x * (blockDim.x >> 5) + (threadIdx.x >> 5);
  // no early return (workgroup barriers below); excess waves redo the last strip (idempotent)
  if (tr >= N_NODES / 16) tr = N_NODES / 16 - 1;
  const int rm = lane & 15;
  const int hi = lane >> 4;

  const float* arow = A + (size_t)(tr * 16 + rm) * K + hi * 8;

  v8f acc[8];
#pragma unroll
  for (int t = 0; t < 8; ++t) acc[t] = {};

  for (int k0 = 0; k0 < K; k0 += KC) {
    if (k0) __syncthreads();              // previous chunk fully consumed before restage
    // ---- stage B chunk: 128 rows x 64 halfwords, via async global->LDS (ASYNCcnt) ----
#pragma unroll
    for (int it = 0; it < 8; ++it) {
      int idx = it * 128 + threadIdx.x;   // 0..1023 -> (row 0..127, seg 0..7), 16B per transfer
      int row = idx >> 3, seg = idx & 7;
      const unsigned short* gp = Bts + (size_t)row * K + k0 + seg * 8;
      unsigned ldsoff = (unsigned)(size_t)(lds_us*)&sB[row * KP + seg * 8];
      asm volatile("global_load_async_to_lds_b128 %0, %1, off"
                   :: "v"(ldsoff), "v"(gp) : "memory");
    }
    asm volatile("s_wait_asynccnt 0x0" ::: "memory");
    __syncthreads();

#pragma unroll
    for (int kk = 0; kk < KC; kk += 32) {
      // A fragment: lane = row m, halfword j -> k = (j/8)*16 + hi*8 + (j%8)
      const float* ap = arow + k0 + kk;
      float4 a0 = *(const float4*)(ap);
      float4 a1 = *(const float4*)(ap + 4);
      float4 a2 = *(const float4*)(ap + 16);
      float4 a3 = *(const float4*)(ap + 20);
      if (relu_in) {
        a0.x = fmaxf(a0.x, 0.f); a0.y = fmaxf(a0.y, 0.f); a0.z = fmaxf(a0.z, 0.f); a0.w = fmaxf(a0.w, 0.f);
        a1.x = fmaxf(a1.x, 0.f); a1.y = fmaxf(a1.y, 0.f); a1.z = fmaxf(a1.z, 0.f); a1.w = fmaxf(a1.w, 0.f);
        a2.x = fmaxf(a2.x, 0.f); a2.y = fmaxf(a2.y, 0.f); a2.z = fmaxf(a2.z, 0.f); a2.w = fmaxf(a2.w, 0.f);
        a3.x = fmaxf(a3.x, 0.f); a3.y = fmaxf(a3.y, 0.f); a3.z = fmaxf(a3.z, 0.f); a3.w = fmaxf(a3.w, 0.f);
      }
      v16us au;
      au[0]  = f2bf(a0.x); au[1]  = f2bf(a0.y); au[2]  = f2bf(a0.z); au[3]  = f2bf(a0.w);
      au[4]  = f2bf(a1.x); au[5]  = f2bf(a1.y); au[6]  = f2bf(a1.z); au[7]  = f2bf(a1.w);
      au[8]  = f2bf(a2.x); au[9]  = f2bf(a2.y); au[10] = f2bf(a2.z); au[11] = f2bf(a2.w);
      au[12] = f2bf(a3.x); au[13] = f2bf(a3.y); au[14] = f2bf(a3.z); au[15] = f2bf(a3.w);
      const v16bf av = __builtin_bit_cast(v16bf, au);

#pragma unroll
      for (int t = 0; t < 8; ++t) {
        // B fragment for column tile t from LDS (rows bank-spread by the +8 pad)
        const unsigned short* brow = &sB[(t * 16 + rm) * KP + kk + hi * 8];
        v8us b0 = *(const v8us*)(brow);
        v8us b1 = *(const v8us*)(brow + 16);
        v16us bu;
#pragma unroll
        for (int i = 0; i < 8; ++i) { bu[i] = b0[i]; bu[i + 8] = b1[i]; }
        acc[t] = __builtin_amdgcn_wmma_f32_16x16x32_bf16(
            false, av, false, __builtin_bit_cast(v16bf, bu),
            (short)0, acc[t], false, false);
      }
    }
  }

  // Epilogue. D layout: VGPR r -> (m = 16*tr + 8*hi + r, n = col0 + 16*t + rm)
  const int mb = tr * 16 + hi * 8;
  float dv2[8];
#pragma unroll
  for (int r = 0; r < 8; ++r) { float d = dinv[mb + r]; dv2[r] = d * d; }
#pragma unroll
  for (int t = 0; t < 8; ++t) {
    const int col = col0 + t * 16 + rm;
    const float bc = bias[col];
    float* xwrow = XW  + (size_t)mb * NTOT + col;
    float* orow  = OUT + (size_t)mb * NTOT + col;
#pragma unroll
    for (int r = 0; r < 8; ++r) {
      float v = acc[t][r];
      xwrow[(size_t)r * NTOT] = v;
      orow[(size_t)r * NTOT]  = dv2[r] * v + bc;
    }
  }
}

// ---------------- edge scatter: out[dst[e]][c] += norm[e] * xw[src[e]][c] ----------------
__global__ void edge_scatter(const int* __restrict__ src, const int* __restrict__ dst,
                             const float* __restrict__ nrm, const float* __restrict__ xw,
                             float* __restrict__ out, int E, int C4, int C) {
  int t = blockIdx.x * blockDim.x + threadIdx.x;
  if (t >= E * C4) return;
  int e  = t / C4;
  int c4 = t - e * C4;
  float w = nrm[e];
  int s = src[e], d = dst[e];
  float4 v = ((const float4*)(xw + (size_t)s * C))[c4];
  float* o = out + (size_t)d * C + c4 * 4;
  atomicAdd(o + 0, w * v.x);
  atomicAdd(o + 1, w * v.y);
  atomicAdd(o + 2, w * v.z);
  atomicAdd(o + 3, w * v.w);
}

// g[batch[n]][c] = max over nodes of relu(h[n][c]); post-relu >= 0 so uint-bit order == float order
__global__ void pool_max(const float* __restrict__ h, const int* __restrict__ batch,
                         float* __restrict__ g, int n, int C) {
  int i = blockIdx.x * blockDim.x + threadIdx.x;
  if (i >= n * C) return;
  int node = i / C;
  int c    = i - node * C;
  float v = fmaxf(h[i], 0.f);
  atomicMax((unsigned int*)&g[(size_t)batch[node] * C + c], __float_as_uint(v));
}

// ---------------- tiny MLP head: one block, LDS ping-pong ----------------
__global__ void mlp_head(const float* __restrict__ g,
                         const float* __restrict__ Wl1, const float* __restrict__ bl1,
                         const float* __restrict__ Wl2, const float* __restrict__ bl2,
                         const float* __restrict__ Wl3, const float* __restrict__ bl3,
                         const float* __restrict__ Wl4, const float* __restrict__ bl4,
                         float* __restrict__ out) {
  __shared__ float s1[N_GRAPHS * 128];   // 32 KB
  __shared__ float s2[N_GRAPHS * 64];    // 16 KB
  const int tid = threadIdx.x;

  for (int idx = tid; idx < N_GRAPHS * 128; idx += blockDim.x) {
    int r = idx >> 7, c = idx & 127;
    float s = bl1[c];
    for (int k = 0; k < 128; ++k) s += g[r * 128 + k] * Wl1[k * 128 + c];
    s1[idx] = fmaxf(s, 0.f);
  }
  __syncthreads();
  for (int idx = tid; idx < N_GRAPHS * 64; idx += blockDim.x) {
    int r = idx >> 6, c = idx & 63;
    float s = bl2[c];
    for (int k = 0; k < 128; ++k) s += s1[r * 128 + k] * Wl2[k * 64 + c];
    s2[idx] = fmaxf(s, 0.f);
  }
  __syncthreads();
  for (int idx = tid; idx < N_GRAPHS * 64; idx += blockDim.x) {
    int r = idx >> 6, c = idx & 63;
    float s = bl3[c];
    for (int k = 0; k < 64; ++k) s += s2[r * 64 + k] * Wl3[k * 64 + c];
    s1[idx] = fmaxf(s, 0.f);
  }
  __syncthreads();
  if (tid < N_GRAPHS) {
    float s = bl4[0];
    for (int k = 0; k < 64; ++k) s += s1[tid * 64 + k] * Wl4[k];
    out[tid] = s;
  }
}

// ---------------- launch ----------------
extern "C" void kernel_launch(void* const* d_in, const int* in_sizes, int n_in,
                              void* d_out, int out_size, void* d_ws, size_t ws_size,
                              hipStream_t stream) {
  (void)in_sizes; (void)n_in; (void)out_size; (void)ws_size;
  const float* x     = (const float*)d_in[0];
  const int*   src   = (const int*)  d_in[1];
  const int*   dst   = (const int*)  d_in[2];
  const float* ew    = (const float*)d_in[3];
  const int*   batch = (const int*)  d_in[4];
  const float* W1 = (const float*)d_in[5];  const float* b1 = (const float*)d_in[6];
  const float* W2 = (const float*)d_in[7];  const float* b2 = (const float*)d_in[8];
  const float* W3 = (const float*)d_in[9];  const float* b3 = (const float*)d_in[10];
  const float* W4 = (const float*)d_in[11]; const float* b4 = (const float*)d_in[12];
  const float* Wl1 = (const float*)d_in[13]; const float* bl1 = (const float*)d_in[14];
  const float* Wl2 = (const float*)d_in[15]; const float* bl2 = (const float*)d_in[16];
  const float* Wl3 = (const float*)d_in[17]; const float* bl3 = (const float*)d_in[18];
  const float* Wl4 = (const float*)d_in[19]; const float* bl4 = (const float*)d_in[20];

  char* ws = (char*)d_ws;
  size_t off = 0;
  auto bump = [&](size_t bytes) -> void* {
    void* p = ws + off;
    off = (off + bytes + 255) & ~(size_t)255;
    return p;
  };
  float* dinv = (float*)bump((size_t)N_NODES * 4);
  float* nrm  = (float*)bump((size_t)N_EDGES * 4);
  float* XW   = (float*)bump((size_t)N_NODES * 256 * 4);
  float* BA   = (float*)bump((size_t)N_NODES * 256 * 4);
  float* BB   = (float*)bump((size_t)N_NODES * 128 * 4);
  float* g    = (float*)bump((size_t)N_GRAPHS * 128 * 4);
  unsigned short* W1t = (unsigned short*)bump((size_t)64 * 256 * 2);
  unsigned short* W2t = (unsigned short*)bump((size_t)256 * 128 * 2);
  unsigned short* W3t = (unsigned short*)bump((size_t)128 * 128 * 2);
  unsigned short* W4t = (unsigned short*)bump((size_t)128 * 128 * 2);

  const int T = 256;
  // degree / norm
  fill_f32<<<(N_NODES + T - 1) / T, T, 0, stream>>>(dinv, 1.0f, N_NODES);
  deg_acc <<<(N_EDGES + T - 1) / T, T, 0, stream>>>(dst, ew, dinv, N_EDGES);
  rsqrt_ip<<<(N_NODES + T - 1) / T, T, 0, stream>>>(dinv, N_NODES);
  norm_k  <<<(N_EDGES + T - 1) / T, T, 0, stream>>>(src, dst, ew, dinv, nrm, N_EDGES);
  // bf16-transposed weights
  conv_wt<<<(64 * 256 + T - 1) / T, T, 0, stream>>>(W1, W1t, 64, 256);
  conv_wt<<<(256 * 128 + T - 1) / T, T, 0, stream>>>(W2, W2t, 256, 128);
  conv_wt<<<(128 * 128 + T - 1) / T, T, 0, stream>>>(W3, W3t, 128, 128);
  conv_wt<<<(128 * 128 + T - 1) / T, T, 0, stream>>>(W4, W4t, 128, 128);

  const int NTR    = N_NODES / 16;           // 3125 row strips
  const int GBLK   = 128;                    // 4 waves per GEMM block
  const int gemm_b = (NTR + 3) / 4;

  // conv1: 64 -> 256, two column halves (input has no relu)
  gemm_wmma_strip<64, 256><<<gemm_b, GBLK, 0, stream>>>(x, W1t,            XW, BA, dinv, b1, 0,   0);
  gemm_wmma_strip<64, 256><<<gemm_b, GBLK, 0, stream>>>(x, W1t + 128 * 64, XW, BA, dinv, b1, 128, 0);
  edge_scatter<<<(N_EDGES * 64 + T - 1) / T, T, 0, stream>>>(src, dst, nrm, XW, BA,
                                                             N_EDGES, 64, 256);
  // conv2: 256 -> 128 (relu fused into A-load)
  gemm_wmma_strip<256, 128><<<gemm_b, GBLK, 0, stream>>>(BA, W2t, XW, BB, dinv, b2, 0, 1);
  edge_scatter<<<(N_EDGES * 32 + T - 1) / T, T, 0, stream>>>(src, dst, nrm, XW, BB,
                                                             N_EDGES, 32, 128);
  // conv3: 128 -> 128
  gemm_wmma_strip<128, 128><<<gemm_b, GBLK, 0, stream>>>(BB, W3t, XW, BA, dinv, b3, 0, 1);
  edge_scatter<<<(N_EDGES * 32 + T - 1) / T, T, 0, stream>>>(src, dst, nrm, XW, BA,
                                                             N_EDGES, 32, 128);
  // conv4: 128 -> 128
  gemm_wmma_strip<128, 128><<<gemm_b, GBLK, 0, stream>>>(BA, W4t, XW, BB, dinv, b4, 0, 1);
  edge_scatter<<<(N_EDGES * 32 + T - 1) / T, T, 0, stream>>>(src, dst, nrm, XW, BB,
                                                             N_EDGES, 32, 128);

  // global max pool (relu fused)
  fill_f32<<<(N_GRAPHS * 128 + T - 1) / T, T, 0, stream>>>(g, 0.0f, N_GRAPHS * 128);
  pool_max<<<(N_NODES * 128 + T - 1) / T, T, 0, stream>>>(BB, batch, g, N_NODES, 128);

  // MLP head
  mlp_head<<<1, 256, 0, stream>>>(g, Wl1, bl1, Wl2, bl2, Wl3, bl3, Wl4, bl4, (float*)d_out);
}
